// CasualSelfAttention_16947940950852
// MI455X (gfx1250) — matrix operand order repaired and checked by reading, not compiled, not run on hardware
//
#include <hip/hip_runtime.h>

typedef __attribute__((ext_vector_type(16))) __bf16 v16bf;
typedef __attribute__((ext_vector_type(8)))  __bf16 v8bf;
typedef __attribute__((ext_vector_type(8)))  float  v8f;
typedef int v4i_vs __attribute__((vector_size(16)));

union BF16x16 { v16bf v; v8bf h[2]; };

// ---- CDNA5 async global->LDS copy (ASYNCcnt path), guarded fallback ------
#if defined(__has_builtin)
# if __has_builtin(__builtin_amdgcn_global_load_async_to_lds_b128) && \
     __has_builtin(__builtin_amdgcn_s_wait_asynccnt)
#  define HAVE_ASYNC_COPY 1
# endif
#endif
#ifndef HAVE_ASYNC_COPY
# define HAVE_ASYNC_COPY 0
#endif

__device__ __forceinline__ void async_copy16(__bf16* ldsDst, const __bf16* gSrc) {
#if HAVE_ASYNC_COPY
  __builtin_amdgcn_global_load_async_to_lds_b128(
      (__attribute__((address_space(1))) v4i_vs*)(gSrc),
      (__attribute__((address_space(3))) v4i_vs*)(ldsDst), 0, 0);
#else
  *(v8bf*)ldsDst = *(const v8bf*)gSrc;
#endif
}
__device__ __forceinline__ void wait_async() {
#if HAVE_ASYNC_COPY
  __builtin_amdgcn_s_wait_asynccnt(0);
#endif
}

__device__ __forceinline__ v8f wmma_bf16(v16bf a, v16bf b, v8f c) {
  return __builtin_amdgcn_wmma_f32_16x16x32_bf16(false, a, false, b, (short)0, c, false, false);
}

// A-matrix 16x32 bf16 fragment from row-major tile (strideElem mult of 8).
__device__ __forceinline__ v16bf load_frag_a(const __bf16* base, int strideElem) {
  const int lane = threadIdx.x & 31;
  const __bf16* p = base + (lane & 15) * strideElem + (lane >> 4) * 8;
  BF16x16 f;
  f.h[0] = *(const v8bf*)(p);
  f.h[1] = *(const v8bf*)(p + 16);
  return f.v;
}

// B-matrix 32x16 bf16 fragment from N-major (N x K row-major) tile.
__device__ __forceinline__ v16bf load_frag_b(const __bf16* base, int strideElem) {
  const int lane = threadIdx.x & 31;
  const __bf16* p = base + (lane & 15) * strideElem + (lane >> 4) * 8;
  BF16x16 f;
  f.h[0] = *(const v8bf*)(p);
  f.h[1] = *(const v8bf*)(p + 16);
  return f.v;
}

// ---------------------------------------------------------------------------
// Kernel: f32 (rows x cols) -> bf16 transposed (cols x rows)
// ---------------------------------------------------------------------------
__global__ __launch_bounds__(256) void convert_transpose_kernel(
    const float* __restrict__ W, __bf16* __restrict__ Wt, int rows, int cols) {
  __shared__ float tile[32][33];
  const int tx = threadIdx.x & 31;
  const int ty = threadIdx.x >> 5;
  const int c0 = blockIdx.x * 32;
  const int r0 = blockIdx.y * 32;
  for (int i = 0; i < 4; ++i)
    tile[ty + i * 8][tx] = W[(size_t)(r0 + ty + i * 8) * cols + (c0 + tx)];
  __syncthreads();
  for (int i = 0; i < 4; ++i)
    Wt[(size_t)(c0 + ty + i * 8) * rows + (r0 + tx)] = (__bf16)tile[tx][ty + i * 8];
}

// ---------------------------------------------------------------------------
// Kernel: x f32 -> bf16 (flat)
// ---------------------------------------------------------------------------
__global__ __launch_bounds__(256) void convert_x_kernel(
    const float* __restrict__ x, __bf16* __restrict__ xb) {
  const size_t i0 = ((size_t)blockIdx.x * 256 + threadIdx.x) * 8;
  const float4 a = *(const float4*)(x + i0);
  const float4 b = *(const float4*)(x + i0 + 4);
  union { v8bf v; __bf16 e[8]; } u;
  u.e[0] = (__bf16)a.x; u.e[1] = (__bf16)a.y; u.e[2] = (__bf16)a.z; u.e[3] = (__bf16)a.w;
  u.e[4] = (__bf16)b.x; u.e[5] = (__bf16)b.y; u.e[6] = (__bf16)b.z; u.e[7] = (__bf16)b.w;
  *(v8bf*)(xb + i0) = u.v;
}

// ---------------------------------------------------------------------------
// Shared GEMM staging: 128 rows x 64 K bf16 tile, global ld = 2048 elems
// ---------------------------------------------------------------------------
#define LDK 72  // 64 + 8 elems; 144B row stride (mult of 16B)

__device__ __forceinline__ void stage128x64(__bf16* dst, const __bf16* src, int blockRow) {
  const int tid = threadIdx.x;
#pragma unroll
  for (int i = 0; i < 4; ++i) {
    const int lin = tid + i * 256;    // 0..1023 16B chunks
    const int r = lin >> 3;
    const int c = (lin & 7) * 8;
    async_copy16(&dst[r * LDK + c], src + (size_t)(blockRow + r) * 2048 + c);
  }
}

// ---------------------------------------------------------------------------
// Kernel: QKV GEMM (xb @ WqkvT^T) + RoPE + scatter q/k (B,H,T,HD), v^T (B,H,HD,T)
// Block 128x128, BK=64 double-buffered async, 8 waves each 64x32.
// ---------------------------------------------------------------------------
__global__ __launch_bounds__(256) void qkv_rope_kernel(
    const __bf16* __restrict__ xb, const float* __restrict__ rope,
    const __bf16* __restrict__ wt,           // (6144, 2048) bf16 N-major
    __bf16* __restrict__ qbuf, __bf16* __restrict__ kbuf, __bf16* __restrict__ vT) {
  __shared__ __attribute__((aligned(16))) __bf16 As[2][128 * LDK];
  __shared__ __attribute__((aligned(16))) __bf16 Bs[2][128 * LDK];

  const int tid  = threadIdx.x;
  const int lane = tid & 31;
  const int wave = tid >> 5;
  const int waveM = (wave >> 2) * 64;
  const int waveN = (wave & 3) * 32;
  const int blockN = blockIdx.x * 128;     // 0..6143
  const int blockM = blockIdx.y * 128;     // 0..8191

  v8f acc[4][2];
  for (int i = 0; i < 4; ++i)
    for (int j = 0; j < 2; ++j)
      acc[i][j] = (v8f){0.f,0.f,0.f,0.f,0.f,0.f,0.f,0.f};

  stage128x64(As[0], xb, blockM);
  stage128x64(Bs[0], wt, blockN);

  for (int it = 0; it < 32; ++it) {
    wait_async();
    __syncthreads();
    if (it < 31) {
      const int kk = (it + 1) * 64;
      stage128x64(As[(it + 1) & 1], xb + kk, blockM);
      stage128x64(Bs[(it + 1) & 1], wt + kk, blockN);
    }
    const __bf16* a = As[it & 1];
    const __bf16* b = Bs[it & 1];
#pragma unroll
    for (int ks = 0; ks < 2; ++ks) {
      v16bf af[4], bfr[2];
      for (int fm = 0; fm < 4; ++fm) af[fm]  = load_frag_a(&a[(waveM + fm * 16) * LDK + ks * 32], LDK);
      for (int fn = 0; fn < 2; ++fn) bfr[fn] = load_frag_b(&b[(waveN + fn * 16) * LDK + ks * 32], LDK);
      for (int fm = 0; fm < 4; ++fm)
        for (int fn = 0; fn < 2; ++fn)
          acc[fm][fn] = wmma_bf16(af[fm], bfr[fn], acc[fm][fn]);
    }
  }

  // epilogue: rope + scatter
  const int section = blockN >> 11;            // 0=q 1=k 2=v
  const int nIn = (blockN & 2047) + waveN;
  const int half = lane >> 4;
  for (int fm = 0; fm < 4; ++fm) {
    const int mbase = blockM + waveM + fm * 16 + 8 * half;
    for (int fn = 0; fn < 2; ++fn) {
      const int col = nIn + fn * 16 + (lane & 15);
      const int h = col >> 7;
      const int d = col & 127;
      for (int r = 0; r < 8; ++r) {
        const int m = mbase + r;
        const int b2 = m >> 11;
        const int t = m & 2047;
        const float val = acc[fm][fn][r];
        if (section == 2) {
          vT[((size_t)(b2 * 16 + h) * 128 + d) * 2048 + t] = (__bf16)val;
        } else {
          const float partner = __shfl_xor(val, 1, 32);
          const float2 cs = *(const float2*)(rope + (size_t)t * 128 + (d >> 1) * 2);
          const float o = (d & 1) ? (val * cs.x + partner * cs.y)
                                  : (val * cs.x - partner * cs.y);
          __bf16* dst = (section == 0) ? qbuf : kbuf;
          dst[((size_t)(b2 * 16 + h) * 2048 + t) * 128 + d] = (__bf16)o;
        }
      }
    }
  }
}

// ---------------------------------------------------------------------------
// Kernel: causal flash attention. Block = (b, h, 128 q rows); 8 waves x 16 rows.
// K/V^T tiles double-buffered via async copies.
// ---------------------------------------------------------------------------
#define LQ 136
#define LK 136
#define LV 72
#define LP 72

__global__ __launch_bounds__(256) void attn_kernel(
    const __bf16* __restrict__ qbuf, const __bf16* __restrict__ kbuf,
    const __bf16* __restrict__ vT, __bf16* __restrict__ ybuf) {
  __shared__ __attribute__((aligned(16))) __bf16 Qs[128 * LQ];
  __shared__ __attribute__((aligned(16))) __bf16 Ks[2][64 * LK];
  __shared__ __attribute__((aligned(16))) __bf16 Vs[2][128 * LV];
  __shared__ __attribute__((aligned(16))) __bf16 Ps[8 * 16 * LP];

  const int tid  = threadIdx.x;
  const int lane = tid & 31;
  const int wave = tid >> 5;
  const int qb = blockIdx.x, h = blockIdx.y, b = blockIdx.z;
  const size_t headQK = (size_t)(b * 16 + h) * 2048 * 128;
  const size_t headV  = (size_t)(b * 16 + h) * 128 * 2048;
  const int q0 = qb * 128;
  const int half = lane >> 4;
  const int col  = lane & 15;

  auto stageK = [&](int k0, __bf16* dstK) {
#pragma unroll
    for (int i = 0; i < 4; ++i) {
      const int lin = tid + i * 256;
      const int r = lin >> 4;
      const int c = (lin & 15) * 8;
      async_copy16(&dstK[r * LK + c], kbuf + headQK + (size_t)(k0 + r) * 128 + c);
    }
  };
  auto stageV = [&](int k0, __bf16* dstV) {
#pragma unroll
    for (int i = 0; i < 4; ++i) {
      const int lin = tid + i * 256;
      const int r = lin >> 3;
      const int c = (lin & 7) * 8;
      async_copy16(&dstV[r * LV + c], vT + headV + (size_t)r * 2048 + k0 + c);
    }
  };

  // stage Q once (async, covered by first wait)
#pragma unroll
  for (int i = 0; i < 8; ++i) {
    const int lin = tid + i * 256;
    const int r = lin >> 4;
    const int c = (lin & 15) * 8;
    async_copy16(&Qs[r * LQ + c], qbuf + headQK + (size_t)(q0 + r) * 128 + c);
  }

  float mrow[8], lrow[8];
  v8f o[8];
  for (int r = 0; r < 8; ++r) { mrow[r] = -3.0e38f; lrow[r] = 0.f; }
  for (int f = 0; f < 8; ++f) o[f] = (v8f){0.f,0.f,0.f,0.f,0.f,0.f,0.f,0.f};

  const float scale = 0.08838834764831845f;  // 1/sqrt(128)
  const int qrow_lane = q0 + wave * 16 + 8 * half;
  const int nk = (qb + 1) * 2;               // causal clip: chunks of 64

  stageK(0, Ks[0]);
  stageV(0, Vs[0]);

  for (int kt = 0; kt < nk; ++kt) {
    wait_async();
    __syncthreads();
    if (kt + 1 < nk) {
      stageK((kt + 1) * 64, Ks[(kt + 1) & 1]);
      stageV((kt + 1) * 64, Vs[(kt + 1) & 1]);
    }
    const __bf16* ks_ = Ks[kt & 1];
    const __bf16* vs_ = Vs[kt & 1];
    const int k0 = kt * 64;

    // S = Q K^T (wave: 16 x 64)
    v8f s[4];
    for (int fn = 0; fn < 4; ++fn) s[fn] = (v8f){0.f,0.f,0.f,0.f,0.f,0.f,0.f,0.f};
#pragma unroll
    for (int dk = 0; dk < 4; ++dk) {
      v16bf aq = load_frag_a(&Qs[(wave * 16) * LQ + dk * 32], LQ);
      for (int fn = 0; fn < 4; ++fn) {
        v16bf bk = load_frag_b(&ks_[(fn * 16) * LK + dk * 32], LK);
        s[fn] = wmma_bf16(aq, bk, s[fn]);
      }
    }

    // scale + causal mask
    for (int fn = 0; fn < 4; ++fn) {
      const int kcol = k0 + fn * 16 + col;
      for (int r = 0; r < 8; ++r) {
        float v = s[fn][r] * scale;
        if (kcol > qrow_lane + r) v = -3.0e38f;
        s[fn][r] = v;
      }
    }

    // online softmax (row spans the 16 lanes of a half-wave + 4 frags)
    for (int r = 0; r < 8; ++r) {
      float mx = fmaxf(fmaxf(s[0][r], s[1][r]), fmaxf(s[2][r], s[3][r]));
      for (int off = 8; off >= 1; off >>= 1) mx = fmaxf(mx, __shfl_xor(mx, off, 32));
      const float mnew = fmaxf(mrow[r], mx);
      const float corr = __expf(mrow[r] - mnew);
      float ls = 0.f;
      for (int fn = 0; fn < 4; ++fn) {
        const float p = __expf(s[fn][r] - mnew);
        s[fn][r] = p;
        ls += p;
      }
      for (int off = 8; off >= 1; off >>= 1) ls += __shfl_xor(ls, off, 32);
      lrow[r] = lrow[r] * corr + ls;
      mrow[r] = mnew;
      for (int f = 0; f < 8; ++f) o[f][r] *= corr;
    }

    // write P (bf16) to wave-private LDS; DS ops are in-order within a wave
    __bf16* pw = &Ps[wave * 16 * LP];
    for (int fn = 0; fn < 4; ++fn)
      for (int r = 0; r < 8; ++r)
        pw[(r + 8 * half) * LP + fn * 16 + col] = (__bf16)s[fn][r];

    // O += P @ V
#pragma unroll
    for (int kc = 0; kc < 2; ++kc) {
      v16bf ap = load_frag_a(pw + kc * 32, LP);
      for (int fd = 0; fd < 8; ++fd) {
        v16bf bv = load_frag_b(&vs_[(fd * 16) * LV + kc * 32], LV);
        o[fd] = wmma_bf16(ap, bv, o[fd]);
      }
    }
  }

  // normalize + store y (B,T,C) bf16
  for (int r = 0; r < 8; ++r) {
    const float inv = 1.0f / lrow[r];
    const int t = q0 + wave * 16 + 8 * half + r;
    __bf16* yrow = ybuf + ((size_t)b * 2048 + t) * 2048 + h * 128;
    for (int fd = 0; fd < 8; ++fd)
      yrow[fd * 16 + col] = (__bf16)(o[fd][r] * inv);
  }
}

// ---------------------------------------------------------------------------
// Kernel: projection GEMM (y @ Wproj), f32 output
// ---------------------------------------------------------------------------
__global__ __launch_bounds__(256) void proj_kernel(
    const __bf16* __restrict__ y, const __bf16* __restrict__ wt, float* __restrict__ out) {
  __shared__ __attribute__((aligned(16))) __bf16 As[2][128 * LDK];
  __shared__ __attribute__((aligned(16))) __bf16 Bs[2][128 * LDK];

  const int tid  = threadIdx.x;
  const int lane = tid & 31;
  const int wave = tid >> 5;
  const int waveM = (wave >> 2) * 64;
  const int waveN = (wave & 3) * 32;
  const int blockN = blockIdx.x * 128;
  const int blockM = blockIdx.y * 128;

  v8f acc[4][2];
  for (int i = 0; i < 4; ++i)
    for (int j = 0; j < 2; ++j)
      acc[i][j] = (v8f){0.f,0.f,0.f,0.f,0.f,0.f,0.f,0.f};

  stage128x64(As[0], y, blockM);
  stage128x64(Bs[0], wt, blockN);

  for (int it = 0; it < 32; ++it) {
    wait_async();
    __syncthreads();
    if (it < 31) {
      const int kk = (it + 1) * 64;
      stage128x64(As[(it + 1) & 1], y + kk, blockM);
      stage128x64(Bs[(it + 1) & 1], wt + kk, blockN);
    }
    const __bf16* a = As[it & 1];
    const __bf16* b = Bs[it & 1];
#pragma unroll
    for (int ks = 0; ks < 2; ++ks) {
      v16bf af[4], bfr[2];
      for (int fm = 0; fm < 4; ++fm) af[fm]  = load_frag_a(&a[(waveM + fm * 16) * LDK + ks * 32], LDK);
      for (int fn = 0; fn < 2; ++fn) bfr[fn] = load_frag_b(&b[(waveN + fn * 16) * LDK + ks * 32], LDK);
      for (int fm = 0; fm < 4; ++fm)
        for (int fn = 0; fn < 2; ++fn)
          acc[fm][fn] = wmma_bf16(af[fm], bfr[fn], acc[fm][fn]);
    }
  }

  const int half = lane >> 4;
  for (int fm = 0; fm < 4; ++fm) {
    const int mbase = blockM + waveM + fm * 16 + 8 * half;
    for (int fn = 0; fn < 2; ++fn) {
      const int n = blockN + waveN + fn * 16 + (lane & 15);
      for (int r = 0; r < 8; ++r)
        out[(size_t)(mbase + r) * 2048 + n] = acc[fm][fn][r];
    }
  }
}

// ---------------------------------------------------------------------------
extern "C" void kernel_launch(void* const* d_in, const int* in_sizes, int n_in,
                              void* d_out, int out_size, void* d_ws, size_t ws_size,
                              hipStream_t stream) {
  const float* x     = (const float*)d_in[0];   // (4,2048,2048)
  const float* rope  = (const float*)d_in[1];   // (2048,64,2)
  const float* Wqkv  = (const float*)d_in[2];   // (2048,6144)
  const float* Wproj = (const float*)d_in[3];   // (2048,2048)
  float* out = (float*)d_out;

  char* ws = (char*)d_ws;
  __bf16* wqkvT  = (__bf16*)(ws);                      // 6144x2048 bf16 = 24 MB
  __bf16* wprojT = (__bf16*)(ws + 25165824);           // 2048x2048 bf16 =  8 MB
  __bf16* xb     = (__bf16*)(ws + 33554432);           // (B,T,C) bf16   = 32 MB
  __bf16* qbuf   = (__bf16*)(ws + 67108864);           // (B,H,T,HD)     = 32 MB
  __bf16* kbuf   = (__bf16*)(ws + 100663296);          // (B,H,T,HD)     = 32 MB
  __bf16* vT     = (__bf16*)(ws + 134217728);          // (B,H,HD,T)     = 32 MB
  __bf16* ybuf   = (__bf16*)(ws + 167772160);          // (B,T,C)        = 32 MB

  convert_transpose_kernel<<<dim3(192, 64), 256, 0, stream>>>(Wqkv, wqkvT, 2048, 6144);
  convert_transpose_kernel<<<dim3(64, 64), 256, 0, stream>>>(Wproj, wprojT, 2048, 2048);
  convert_x_kernel<<<dim3(8192), 256, 0, stream>>>(x, xb);
  qkv_rope_kernel<<<dim3(48, 64), 256, 0, stream>>>(xb, rope, wqkvT, qbuf, kbuf, vT);
  attn_kernel<<<dim3(16, 16, 4), 256, 0, stream>>>(qbuf, kbuf, vT, ybuf);
  proj_kernel<<<dim3(16, 64), 256, 0, stream>>>(ybuf, wprojT, out);
}
